// CausalSelfAttention_8899172237349
// MI455X (gfx1250) — matrix operand order repaired and checked by reading, not compiled, not run on hardware
//
#include <hip/hip_runtime.h>

#define T_SEQ 2048
#define C_EMB 1024
#define H_NUM 16
#define H_DIM 64
#define B_NUM 4

typedef __attribute__((ext_vector_type(16))) __bf16 v16bf;
typedef __attribute__((ext_vector_type(8)))  float  v8f;
typedef __attribute__((ext_vector_type(4)))  unsigned int v4u;
typedef __attribute__((ext_vector_type(8)))  unsigned int v8u;
typedef int v4i __attribute__((vector_size(16)));  // matches builtin's V4i

struct Frag {
  union { unsigned int u[8]; v16bf h; };
};

__device__ __forceinline__ unsigned short f2bf(float f) {
  unsigned int u = __float_as_uint(f);
  u += 0x7FFFu + ((u >> 16) & 1u);
  return (unsigned short)(u >> 16);
}

// --- CDNA5 async / tensor plumbing --------------------------------------
__device__ __forceinline__ void wait_asynccnt0() {
#if __has_builtin(__builtin_amdgcn_s_wait_asynccnt)
  __builtin_amdgcn_s_wait_asynccnt(0);
#else
  asm volatile("s_wait_asynccnt 0x0" ::: "memory");
#endif
}

__device__ __forceinline__ void wait_tensorcnt0() {
#if __has_builtin(__builtin_amdgcn_s_wait_tensorcnt)
  __builtin_amdgcn_s_wait_tensorcnt(0);
#else
  asm volatile("s_wait_tensorcnt 0x0" ::: "memory");
#endif
}

__device__ __forceinline__ unsigned lds_off(const void* p) {
  return (unsigned)(unsigned long long)
      (__attribute__((address_space(3))) const void*)p;
}

// async global -> LDS, 16 bytes per lane (ASYNCcnt-tracked, no VGPR data)
__device__ __forceinline__ void async_b128(const unsigned short* g,
                                           unsigned short* l) {
#if __has_builtin(__builtin_amdgcn_global_load_async_to_lds_b128)
  void* gv = (void*)g;
  void* lv = (void*)l;
  __builtin_amdgcn_global_load_async_to_lds_b128(
      (__attribute__((address_space(1))) v4i*)gv,
      (__attribute__((address_space(3))) v4i*)lv, 0, 0);
#else
  unsigned loff = lds_off(l);
  asm volatile("global_load_async_to_lds_b128 %0, %1, off"
               :: "v"(loff), "v"(g) : "memory");
#endif
}

// TDM: 2D bf16 tile (tileW x tileH elems) from global into LDS.
// D# built in SGPRs per CDNA5 ISA; inline asm is
// portable across the 5-arg/6-arg builtin arity split.
__device__ __forceinline__ void tdm_load_2d_bf16(const unsigned short* gtile,
                                                 unsigned ldsOff,
                                                 unsigned tileW, unsigned tileH,
                                                 unsigned rowStride,
                                                 unsigned tensorRows) {
  const unsigned long long ga = (unsigned long long)gtile;
  v4u g0;
  g0.x = 1u;                                         // count=1, user mode
  g0.y = ldsOff;                                     // lds_addr (bytes)
  g0.z = (unsigned)ga;                               // global_addr[31:0]
  g0.w = ((unsigned)(ga >> 32) & 0x01FFFFFFu) | 0x80000000u;  // [56:32]|type=2
  v8u g1;
  g1[0] = 0x00010000u;                               // data_size = 2 bytes
  g1[1] = (rowStride & 0xFFFFu) << 16;               // tensor_dim0[15:0]
  g1[2] = ((rowStride >> 16) & 0xFFFFu) |            // tensor_dim0[31:16]
          ((tensorRows & 0xFFFFu) << 16);            // tensor_dim1[15:0]
  g1[3] = ((tensorRows >> 16) & 0xFFFFu) |           // tensor_dim1[31:16]
          ((tileW & 0xFFFFu) << 16);                 // tile_dim0
  g1[4] = tileH & 0xFFFFu;                           // tile_dim1 (tile_dim2=0)
  g1[5] = rowStride;                                 // tensor_dim0_stride[31:0]
  g1[6] = 0u;
  g1[7] = 0u;
  asm volatile("tensor_load_to_lds %0, %1" :: "s"(g0), "s"(g1) : "memory");
}

// ---------------------------------------------------------------------------
// f32 -> bf16 pack (memory bound)
// ---------------------------------------------------------------------------
__global__ void cvt_bf16_kernel(const float* __restrict__ in,
                                unsigned short* __restrict__ out, int n) {
  int i = blockIdx.x * blockDim.x + threadIdx.x;
  int stride = gridDim.x * blockDim.x;
  for (; i < n; i += stride) out[i] = f2bf(in[i]);
}

// ---------------------------------------------------------------------------
// Tiled bf16 WMMA GEMM: C = A[M,K] * B[K,N] + bias
// Block tile 128x128, K-step 32, 8 waves, wave tile 32x64.
// A tile staged via async global->LDS (double buffered); B tile K-transposed
// through VGPRs into the alternate buffer. One barrier per K-step.
// mode 0: f32 C row-major.  mode 1: scatter bf16 Q/K/V as [B,H,T,64].
// ---------------------------------------------------------------------------
__global__ __launch_bounds__(256)
void gemm_bf16_wmma(const unsigned short* __restrict__ A,
                    const unsigned short* __restrict__ Bw,
                    const float* __restrict__ bias,
                    float* __restrict__ Cf,
                    unsigned short* __restrict__ Qo,
                    unsigned short* __restrict__ Ko,
                    unsigned short* __restrict__ Vo,
                    int M, int N, int K, int mode) {
  __shared__ unsigned short As[2][128 * 32];  // [m][k]
  __shared__ unsigned short Bs[2][128 * 32];  // transposed [n][k]

  const int tid  = threadIdx.x;
  const int lane = tid & 31;
  const int w    = tid >> 5;
  const int l15  = lane & 15;
  const int hh   = lane >> 4;

  const int row0 = blockIdx.y * 128;
  const int col0 = blockIdx.x * 128;
  const int mbase = (w & 3) * 32;
  const int nbase = (w >> 2) * 64;

  v8f acc[2][4];
  for (int a = 0; a < 2; a++)
    for (int b = 0; b < 4; b++)
      for (int j = 0; j < 8; j++) acc[a][b][j] = 0.0f;

  const int ar = tid >> 1, ahalf = tid & 1;  // A tile: 128 rows x 32 k
  const int bk = tid >> 3, bseg  = tid & 7;  // B tile: 32 k x 128 n

  union BReg { uint4 q[2]; unsigned short s[16]; };

  auto issueA = [&](int k0, int buf) {
    const unsigned short* g = A + (size_t)(row0 + ar) * K + k0 + ahalf * 16;
    unsigned short* l = &As[buf][ar * 32 + ahalf * 16];
    async_b128(g, l);
    async_b128(g + 8, l + 8);
  };
  auto loadB = [&](int k0, BReg& r) {
    const uint4* bg = reinterpret_cast<const uint4*>(
        Bw + (size_t)(k0 + bk) * N + col0 + bseg * 16);
    r.q[0] = bg[0];
    r.q[1] = bg[1];
  };
  auto storeB = [&](const BReg& r, int buf) {
#pragma unroll
    for (int e = 0; e < 16; e++)
      Bs[buf][(bseg * 16 + e) * 32 + bk] = r.s[e];  // K-transpose on the fly
  };

  // prologue: stage tile 0
  BReg breg;
  issueA(0, 0);
  loadB(0, breg);
  storeB(breg, 0);
  wait_asynccnt0();
  __syncthreads();

  int cur = 0;
  const int nIter = K >> 5;
  for (int it = 0; it < nIter; ++it) {
    const bool hasNext = (it + 1 < nIter);
    if (hasNext) {
      issueA((it + 1) << 5, cur ^ 1);  // async copy overlaps the WMMAs below
      loadB((it + 1) << 5, breg);
    }

    Frag af[2], bf[4];
#pragma unroll
    for (int mt = 0; mt < 2; mt++) {
      const int mrow = mbase + mt * 16 + l15;
#pragma unroll
      for (int v = 0; v < 8; v++) {
        const int kk = ((v >> 2) << 4) + (hh << 3) + ((v & 3) << 1);  // A layout
        af[mt].u[v] =
            *reinterpret_cast<const unsigned int*>(&As[cur][mrow * 32 + kk]);
      }
    }
#pragma unroll
    for (int nt = 0; nt < 4; nt++) {
      const int ncol = nbase + nt * 16 + l15;
#pragma unroll
      for (int v = 0; v < 8; v++) {
        const int kk = ((v >> 2) << 4) + (hh << 3) + ((v & 3) << 1);  // B layout
        bf[nt].u[v] =
            *reinterpret_cast<const unsigned int*>(&Bs[cur][ncol * 32 + kk]);
      }
    }
#pragma unroll
    for (int mt = 0; mt < 2; mt++)
#pragma unroll
      for (int nt = 0; nt < 4; nt++)
        acc[mt][nt] = __builtin_amdgcn_wmma_f32_16x16x32_bf16(
            false, af[mt].h, false, bf[nt].h, (short)0, acc[mt][nt], false, false);

    if (hasNext) storeB(breg, cur ^ 1);  // other buffer: no hazard with readers
    wait_asynccnt0();
    __syncthreads();
    cur ^= 1;
  }

  // epilogue: C layout -> lane holds col l15, rows v + 8*hh
#pragma unroll
  for (int mt = 0; mt < 2; mt++) {
#pragma unroll
    for (int nt = 0; nt < 4; nt++) {
#pragma unroll
      for (int v = 0; v < 8; v++) {
        const int rM = row0 + mbase + mt * 16 + v + (hh << 3);
        const int cN = col0 + nbase + nt * 16 + l15;
        const float val = acc[mt][nt][v] + bias[cN];
        if (mode == 0) {
          Cf[(size_t)rM * N + cN] = val;
        } else {
          const int sel = cN >> 10;  // 0=Q 1=K 2=V
          const int c   = cN & 1023;
          const int hI  = c >> 6, dI = c & 63;
          const int bI  = rM >> 11, tI = rM & 2047;
          const size_t dst =
              (((size_t)(bI * H_NUM + hI) * T_SEQ) + tI) * H_DIM + dI;
          const unsigned short bv = f2bf(val);
          if (sel == 0)      Qo[dst] = bv;
          else if (sel == 1) Ko[dst] = bv;
          else               Vo[dst] = bv;
        }
      }
    }
  }
}

// ---------------------------------------------------------------------------
// Flash attention: grid (T/128, B*H), 8 waves, 16 q-rows/wave.
// K tile (64x64 bf16) staged by the Tensor Data Mover (wave 0 programs D#);
// V staged transposed through VGPRs. Online softmax, WMMA for QK^T and PV.
// ---------------------------------------------------------------------------
__global__ __launch_bounds__(256)
void attn_kernel(const unsigned short* __restrict__ Qw,
                 const unsigned short* __restrict__ Kw,
                 const unsigned short* __restrict__ Vw,
                 unsigned short* __restrict__ Yw) {
  __shared__ unsigned short Ks[64 * 64];     // [key][d]   (TDM target)
  __shared__ unsigned short Vt[64 * 64];     // [d][key]   (transposed)
  __shared__ unsigned short Ps[8][16 * 64];  // per-wave P scratch [row][key]

  const int tid  = threadIdx.x;
  const int lane = tid & 31;
  const int w    = tid >> 5;
  const int l15  = lane & 15;
  const int hh   = lane >> 4;

  const int qblk = blockIdx.x;   // 0..15
  const int bh   = blockIdx.y;   // b*H + h, 0..63
  const size_t base = (size_t)bh * T_SEQ * H_DIM;
  const int tRow0 = qblk * 128 + w * 16;

  // Q fragments (A layout), 2 k-steps over D=64, loaded once, reused
  Frag aq[2];
#pragma unroll
  for (int ks = 0; ks < 2; ks++) {
    const size_t qrow = base + (size_t)(tRow0 + l15) * H_DIM;
#pragma unroll
    for (int v = 0; v < 8; v++) {
      const int kk = ks * 32 + ((v >> 2) << 4) + (hh << 3) + ((v & 3) << 1);
      aq[ks].u[v] = *reinterpret_cast<const unsigned int*>(&Qw[qrow + kk]);
    }
  }

  v8f o[4];
  for (int dt = 0; dt < 4; dt++)
    for (int j = 0; j < 8; j++) o[dt][j] = 0.0f;
  float mrow[8], lrow[8];
#pragma unroll
  for (int v = 0; v < 8; v++) { mrow[v] = -1e30f; lrow[v] = 0.0f; }

  const int kbMax = qblk * 2 + 1;
  const int lr = tid >> 2, lseg = tid & 3;

  for (int kb = 0; kb <= kbMax; kb++) {
    // V into registers (overlaps tail of previous iteration's compute)
    union { uint4 q[2]; unsigned short s[16]; } vld;
    const uint4* vg = reinterpret_cast<const uint4*>(
        &Vw[base + (size_t)(kb * 64 + lr) * H_DIM + lseg * 16]);
    vld.q[0] = vg[0]; vld.q[1] = vg[1];

    __syncthreads();  // previous iteration done reading Ks/Vt

    if (tid < 32) {  // wave 0 drives the TDM for the K tile
      tdm_load_2d_bf16(&Kw[base + (size_t)kb * 64 * H_DIM], lds_off(Ks),
                       /*tileW=*/H_DIM, /*tileH=*/64,
                       /*rowStride=*/H_DIM, /*tensorRows=*/T_SEQ);
      wait_tensorcnt0();
    }
#pragma unroll
    for (int e = 0; e < 16; e++)
      Vt[(lseg * 16 + e) * 64 + lr] = vld.s[e];  // transpose V into LDS
    __syncthreads();

    // S = Q * K^T
    v8f s[4];
    for (int nt = 0; nt < 4; nt++)
      for (int j = 0; j < 8; j++) s[nt][j] = 0.0f;
#pragma unroll
    for (int ks = 0; ks < 2; ks++) {
#pragma unroll
      for (int nt = 0; nt < 4; nt++) {
        Frag bk;
        const int krow = nt * 16 + l15;
#pragma unroll
        for (int v = 0; v < 8; v++) {
          const int kk = ks * 32 + ((v >> 2) << 4) + (hh << 3) + ((v & 3) << 1);
          bk.u[v] = *reinterpret_cast<const unsigned int*>(&Ks[krow * 64 + kk]);
        }
        s[nt] = __builtin_amdgcn_wmma_f32_16x16x32_bf16(
            false, aq[ks].h, false, bk.h, (short)0, s[nt], false, false);
      }
    }

    // scale 1/sqrt(64), causal mask, online softmax (16-lane row groups)
    float p[4][8];
#pragma unroll
    for (int v = 0; v < 8; v++) {
      const int rG = tRow0 + v + (hh << 3);
      float mx = -1e30f;
#pragma unroll
      for (int nt = 0; nt < 4; nt++) {
        const int cG = kb * 64 + nt * 16 + l15;
        float sv = s[nt][v] * 0.125f;
        if (cG > rG) sv = -1e30f;
        p[nt][v] = sv;
        mx = fmaxf(mx, sv);
      }
      mx = fmaxf(mx, __shfl_xor(mx, 1, 16));
      mx = fmaxf(mx, __shfl_xor(mx, 2, 16));
      mx = fmaxf(mx, __shfl_xor(mx, 4, 16));
      mx = fmaxf(mx, __shfl_xor(mx, 8, 16));
      const float mnew  = fmaxf(mrow[v], mx);
      const float alpha = __expf(mrow[v] - mnew);
      float lsum = 0.0f;
#pragma unroll
      for (int nt = 0; nt < 4; nt++) {
        const float e = __expf(p[nt][v] - mnew);
        p[nt][v] = e;
        lsum += e;
      }
      lsum += __shfl_xor(lsum, 1, 16);
      lsum += __shfl_xor(lsum, 2, 16);
      lsum += __shfl_xor(lsum, 4, 16);
      lsum += __shfl_xor(lsum, 8, 16);
      lrow[v] = lrow[v] * alpha + lsum;
      mrow[v] = mnew;
#pragma unroll
      for (int dt = 0; dt < 4; dt++) o[dt][v] *= alpha;
#pragma unroll
      for (int nt = 0; nt < 4; nt++)
        Ps[w][(v + (hh << 3)) * 64 + nt * 16 + l15] = f2bf(p[nt][v]);
    }

    // O += P * V  (per-wave LDS round trip; same-wave DS ops are in-order)
#pragma unroll
    for (int ks = 0; ks < 2; ks++) {
      Frag ap;
#pragma unroll
      for (int v = 0; v < 8; v++) {
        const int kk = ks * 32 + ((v >> 2) << 4) + (hh << 3) + ((v & 3) << 1);
        ap.u[v] = *reinterpret_cast<const unsigned int*>(&Ps[w][l15 * 64 + kk]);
      }
#pragma unroll
      for (int dt = 0; dt < 4; dt++) {
        Frag bv;
        const int dcol = dt * 16 + l15;
#pragma unroll
        for (int v = 0; v < 8; v++) {
          const int kk = ks * 32 + ((v >> 2) << 4) + (hh << 3) + ((v & 3) << 1);
          bv.u[v] = *reinterpret_cast<const unsigned int*>(&Vt[dcol * 64 + kk]);
        }
        o[dt] = __builtin_amdgcn_wmma_f32_16x16x32_bf16(
            false, ap.h, false, bv.h, (short)0, o[dt], false, false);
      }
    }
  }
  __syncthreads();

  // normalize and write bf16 output directly in [B, T, C] layout
  const int bB = bh >> 4, hH = bh & 15;
#pragma unroll
  for (int v = 0; v < 8; v++) {
    const int tG = tRow0 + v + (hh << 3);
    const float inv = 1.0f / lrow[v];
#pragma unroll
    for (int dt = 0; dt < 4; dt++) {
      const int cG = hH * H_DIM + dt * 16 + l15;
      Yw[((size_t)(bB * T_SEQ + tG)) * C_EMB + cG] = f2bf(o[dt][v] * inv);
    }
  }
}

// ---------------------------------------------------------------------------
extern "C" void kernel_launch(void* const* d_in, const int* in_sizes, int n_in,
                              void* d_out, int out_size, void* d_ws, size_t ws_size,
                              hipStream_t stream) {
  (void)in_sizes; (void)n_in; (void)out_size; (void)ws_size;
  const float* x      = (const float*)d_in[0];
  const float* w_attn = (const float*)d_in[1];
  const float* b_attn = (const float*)d_in[2];
  const float* w_proj = (const float*)d_in[3];
  const float* b_proj = (const float*)d_in[4];
  float* out = (float*)d_out;

  const size_t M = (size_t)B_NUM * T_SEQ;  // 8192
  char* ws = (char*)d_ws;
  unsigned short* xh  = (unsigned short*)ws; ws += M * C_EMB * 2;
  unsigned short* wah = (unsigned short*)ws; ws += (size_t)C_EMB * 3 * C_EMB * 2;
  unsigned short* wph = (unsigned short*)ws; ws += (size_t)C_EMB * C_EMB * 2;
  unsigned short* qw  = (unsigned short*)ws; ws += M * C_EMB * 2;
  unsigned short* kw  = (unsigned short*)ws; ws += M * C_EMB * 2;
  unsigned short* vw  = (unsigned short*)ws; ws += M * C_EMB * 2;
  unsigned short* yw  = (unsigned short*)ws; ws += M * C_EMB * 2;

  cvt_bf16_kernel<<<4096, 256, 0, stream>>>(x, xh, (int)(M * C_EMB));
  cvt_bf16_kernel<<<4096, 256, 0, stream>>>(w_attn, wah, C_EMB * 3 * C_EMB);
  cvt_bf16_kernel<<<2048, 256, 0, stream>>>(w_proj, wph, C_EMB * C_EMB);

  dim3 g1(3 * C_EMB / 128, (unsigned)(M / 128));
  gemm_bf16_wmma<<<g1, 256, 0, stream>>>(xh, wah, b_attn, nullptr, qw, kw, vw,
                                         (int)M, 3 * C_EMB, C_EMB, 1);

  dim3 g2(T_SEQ / 128, B_NUM * H_NUM);
  attn_kernel<<<g2, 256, 0, stream>>>(qw, kw, vw, yw);

  dim3 g3(C_EMB / 128, (unsigned)(M / 128));
  gemm_bf16_wmma<<<g3, 256, 0, stream>>>(yw, wph, b_proj, out,
                                         nullptr, nullptr, nullptr,
                                         (int)M, C_EMB, C_EMB, 0);
}
